// ChessNNUE_67053029425688
// MI455X (gfx1250) — matrix operand order repaired and checked by reading, not compiled, not run on hardware
//
#include <hip/hip_runtime.h>
#include <cstdint>

typedef _Float16 v16h __attribute__((ext_vector_type(16)));
typedef float    v8f  __attribute__((ext_vector_type(8)));

// async-to-LDS builtin operand types: pointer to GNU vector of 4 ints,
// address_space(1) = global, address_space(3) = LDS
typedef int v4i_g __attribute__((vector_size(16)));
typedef __attribute__((address_space(1))) v4i_g* gv4i_p;
typedef __attribute__((address_space(3))) v4i_g* lv4i_p;

#define K_DIM      768
#define KSTEPS     24                    // 768 / 32
#define FRAG_DW    8                     // dwords per lane per 16x32 f16 fragment
#define NT_FRAG_DW (KSTEPS * 32 * FRAG_DW)  // 6144 dwords per n-tile column of B
#define BGROUP_DW  (4 * NT_FRAG_DW)         // 24576 dwords = 96KB per 64-col group

static __device__ __forceinline__ uint32_t pack_f16(float a, float b) {
  union { _Float16 h[2]; uint32_t u; } t;
  t.h[0] = (_Float16)a; t.h[1] = (_Float16)b;
  return t.u;
}
static __device__ __forceinline__ float clip01(float x) {
  return fminf(fmaxf(x, 0.0f), 1.0f);
}

// ---------------------------------------------------------------------------
// Kernel 1: convert ft_w [1024 x 768] f32 -> f16, pre-swizzled into WMMA-B
// fragment layout: bswz[ntile(64)][kstep(24)][lane(32)][dword i(8)]
//   lane L: column n = ntile*16 + (L&15); dword i holds K pair
//   k = kstep*32 + (i>>2)*16 + (L>>4)*8 + 2*(i&3)   (B rows striped across lanes per VGPR)
// ---------------------------------------------------------------------------
__global__ __launch_bounds__(256) void nnue_prep_ftw(
    const float* __restrict__ ftw, uint32_t* __restrict__ bswz) {
  uint32_t idx  = blockIdx.x * 256u + threadIdx.x;   // < 64*24*32*8 = 393216
  uint32_t i    = idx & 7u;
  uint32_t lane = (idx >> 3) & 31u;
  uint32_t rest = idx >> 8;
  uint32_t kstep = rest % 24u;
  uint32_t ntile = rest / 24u;
  uint32_t n = ntile * 16u + (lane & 15u);
  uint32_t k = kstep * 32u + (i >> 2) * 16u + (lane >> 4) * 8u + (i & 3u) * 2u;
  const float* p = ftw + (size_t)n * K_DIM + k;
  bswz[idx] = pack_f16(p[0], p[1]);
}

// ---------------------------------------------------------------------------
// Kernel 2: fused NNUE forward. One block = 64 batch samples.
//   LDS A panel: 128 rows (64 white + 64 black) x 768 as f16 A-fragments
//   Loop over 16 groups of 64 H-columns: WMMA, stm-mix+clip, L1 partials.
//   Finish with L2/L3/sigmoid.
// ---------------------------------------------------------------------------
__global__ __launch_bounds__(256) void nnue_main(
    const float* __restrict__ wf, const float* __restrict__ bfeat,
    const float* __restrict__ stm, const uint32_t* __restrict__ bswz,
    const float* __restrict__ ftb, const float* __restrict__ l1w,
    const float* __restrict__ l1b, const float* __restrict__ l2w,
    const float* __restrict__ l2b, const float* __restrict__ l3w,
    const float* __restrict__ l3b, float* __restrict__ out, int nTotal) {
  // 8 m-tiles * 24 ksteps * 32 lanes * 8 dwords = 49152 dwords = 192KB
  __shared__ __attribute__((aligned(32))) uint32_t sA[8 * KSTEPS * 32 * FRAG_DW];
  // B panel for one 64-column group: 96KB (re-used as f32 acc staging)
  __shared__ __attribute__((aligned(32))) uint32_t sB[BGROUP_DW];
  float* sAcc = (float*)sB;   // [128][65] padded rows (33,280 B < 96KB)

  const int t    = threadIdx.x;
  const int lane = t & 31;
  const int wav  = __builtin_amdgcn_readfirstlane(t >> 5);   // 0..7
  const int batch0 = blockIdx.x * 64;

  // ---- Phase 1: stage A panel (rows 0..63 white, 64..127 black) as f16 frags
  for (int idx = t; idx < 128 * 384; idx += 256) {
    int row = idx / 384;
    int j   = idx - row * 384;
    int k   = j * 2;
    const float* src = (row < 64)
        ? (wf    + (size_t)(batch0 + row)      * K_DIM)
        : (bfeat + (size_t)(batch0 + row - 64) * K_DIM);
    float f0 = src[k], f1 = src[k + 1];
    // invert A-fragment layout: lanes 0-15 K{0..7,16..23}, 16-31 K{8..15,24..31}
    int mtile = row >> 4, l15 = row & 15;
    int kstep = k >> 5,   k32 = k & 31;
    int hi    = k32 >> 4;
    int rem   = k32 & 15;
    int khalf = rem >> 3;
    int i     = hi * 4 + ((rem & 7) >> 1);
    int dst   = ((mtile * KSTEPS + kstep) * 32 + (khalf * 16 + l15)) * FRAG_DW + i;
    sA[dst] = pack_f16(f0, f1);
  }

  // per-thread L1 partial accumulators: this thread covers samples lane and
  // lane+32 over columns {wav*8 .. wav*8+7} of every 64-column group.
  float p0[8], p1[8];
#pragma unroll
  for (int j = 0; j < 8; ++j) { p0[j] = 0.0f; p1[j] = 0.0f; }
  const float stm0 = stm[batch0 + lane];
  const float stm1 = stm[batch0 + lane + 32];

  __syncthreads();

  for (int ng = 0; ng < 16; ++ng) {
    // ---- stage B panel (pre-swizzled f16, contiguous 96KB) into LDS
#if __has_builtin(__builtin_amdgcn_global_load_async_to_lds_b128)
    {
      const uint32_t* gsrc = bswz + (size_t)ng * BGROUP_DW;
      for (int idx = t; idx < BGROUP_DW / 4; idx += 256) {
        __builtin_amdgcn_global_load_async_to_lds_b128(
            (gv4i_p)(unsigned long long)(gsrc + idx * 4),
            (lv4i_p)(unsigned int)(unsigned long long)(sB + idx * 4),
            0, 0);
      }
#if __has_builtin(__builtin_amdgcn_s_wait_asynccnt)
      __builtin_amdgcn_s_wait_asynccnt(0);
#else
      asm volatile("s_wait_asynccnt 0" ::: "memory");
#endif
    }
#else
    {
      const uint4* gsrc = (const uint4*)(bswz + (size_t)ng * BGROUP_DW);
      uint4* ldst = (uint4*)sB;
      for (int idx = t; idx < BGROUP_DW / 4; idx += 256) ldst[idx] = gsrc[idx];
    }
#endif
    __syncthreads();

    // ---- WMMA: wave `wav` computes rows [wav*16, wav*16+16) x 64 cols
    v8f acc0 = {}, acc1 = {}, acc2 = {}, acc3 = {};
    const uint32_t* aBase = sA + (wav * KSTEPS) * 32 * FRAG_DW;
#pragma unroll 6
    for (int k = 0; k < KSTEPS; ++k) {
      v16h a  = *(const v16h*)(aBase + (k * 32 + lane) * FRAG_DW);
      v16h b0 = *(const v16h*)(sB + ((0 * KSTEPS + k) * 32 + lane) * FRAG_DW);
      v16h b1 = *(const v16h*)(sB + ((1 * KSTEPS + k) * 32 + lane) * FRAG_DW);
      v16h b2 = *(const v16h*)(sB + ((2 * KSTEPS + k) * 32 + lane) * FRAG_DW);
      v16h b3 = *(const v16h*)(sB + ((3 * KSTEPS + k) * 32 + lane) * FRAG_DW);
      acc0 = __builtin_amdgcn_wmma_f32_16x16x32_f16(false, a, false, b0, (short)0, acc0, false, false);
      acc1 = __builtin_amdgcn_wmma_f32_16x16x32_f16(false, a, false, b1, (short)0, acc1, false, false);
      acc2 = __builtin_amdgcn_wmma_f32_16x16x32_f16(false, a, false, b2, (short)0, acc2, false, false);
      acc3 = __builtin_amdgcn_wmma_f32_16x16x32_f16(false, a, false, b3, (short)0, acc3, false, false);
    }
    __syncthreads();   // all waves done reading sB -> safe to overwrite as acc

    // ---- write C tiles to padded f32 LDS: row = wav*16 + (lane>>4)*8 + v
    {
      int col0 = lane & 15;
      int rowb = wav * 16 + (lane >> 4) * 8;
#pragma unroll
      for (int v = 0; v < 8; ++v) {
        sAcc[(rowb + v) * 65 + col0 +  0] = acc0[v];
        sAcc[(rowb + v) * 65 + col0 + 16] = acc1[v];
        sAcc[(rowb + v) * 65 + col0 + 32] = acc2[v];
        sAcc[(rowb + v) * 65 + col0 + 48] = acc3[v];
      }
    }
    __syncthreads();

    // ---- fused stm-mix + clip + L1 partials (wave-uniform weight indices)
#pragma unroll
    for (int cc = 0; cc < 8; ++cc) {
      int   c_l = wav * 8 + cc;          // uniform per wave
      int   c_g = ng * 64 + c_l;
      float fb  = ftb[c_g];              // scalar load
      float w0v = sAcc[ lane       * 65 + c_l] + fb;
      float b0v = sAcc[(64 + lane) * 65 + c_l] + fb;
      float w1v = sAcc[(lane + 32) * 65 + c_l] + fb;
      float b1v = sAcc[(96 + lane) * 65 + c_l] + fb;
      float a00 = clip01(stm0 * w0v + (1.0f - stm0) * b0v);  // acc col c_g
      float a10 = clip01(stm0 * b0v + (1.0f - stm0) * w0v);  // acc col 1024+c_g
      float a01 = clip01(stm1 * w1v + (1.0f - stm1) * b1v);
      float a11 = clip01(stm1 * b1v + (1.0f - stm1) * w1v);
#pragma unroll
      for (int j = 0; j < 8; ++j) {
        float wj0 = l1w[j * 2048 + c_g];          // scalar loads (uniform)
        float wj1 = l1w[j * 2048 + 1024 + c_g];
        p0[j] += wj0 * a00 + wj1 * a10;
        p1[j] += wj0 * a01 + wj1 * a11;
      }
    }
    __syncthreads();   // protect sB before next group's staging
  }

  // ---- cross-wave reduction of L1 partials (reuse sA: 8*64*8 f32 = 16KB)
  float* red = (float*)sA;
#pragma unroll
  for (int j = 0; j < 8; ++j) {
    red[(wav * 64 + lane)      * 8 + j] = p0[j];
    red[(wav * 64 + lane + 32) * 8 + j] = p1[j];
  }
  __syncthreads();

  if (t < 64) {
    float h[8];
#pragma unroll
    for (int j = 0; j < 8; ++j) {
      float s = 0.0f;
#pragma unroll
      for (int w = 0; w < 8; ++w) s += red[(w * 64 + t) * 8 + j];
      h[j] = clip01(s + l1b[j]);         // l2_x
    }
    float raw = l3b[0];
#pragma unroll 4
    for (int o = 0; o < 32; ++o) {
      float v = l2b[o];
#pragma unroll
      for (int j = 0; j < 8; ++j) v += l2w[o * 8 + j] * h[j];
      raw += l3w[o] * clip01(v);         // l3_x contribution
    }
    int gs = batch0 + t;
    out[gs]          = 1.0f / (1.0f + __expf(-raw));   // sigmoid output
    out[nTotal + gs] = raw;                            // raw_output
  }
}

extern "C" void kernel_launch(void* const* d_in, const int* in_sizes, int n_in,
                              void* d_out, int out_size, void* d_ws, size_t ws_size,
                              hipStream_t stream) {
  (void)n_in; (void)out_size; (void)ws_size;
  const float* wf  = (const float*)d_in[0];
  const float* bf  = (const float*)d_in[1];
  const float* stm = (const float*)d_in[2];
  const float* ftw = (const float*)d_in[3];
  const float* ftb = (const float*)d_in[4];
  const float* l1w = (const float*)d_in[5];
  const float* l1b = (const float*)d_in[6];
  const float* l2w = (const float*)d_in[7];
  const float* l2b = (const float*)d_in[8];
  const float* l3w = (const float*)d_in[9];
  const float* l3b = (const float*)d_in[10];
  float* out = (float*)d_out;
  uint32_t* bswz = (uint32_t*)d_ws;    // 64*24*32*8 dwords = 1.5MB

  const int B = in_sizes[0] / K_DIM;   // 65536

  nnue_prep_ftw<<<(64 * KSTEPS * 32 * FRAG_DW) / 256, 256, 0, stream>>>(ftw, bswz);
  nnue_main<<<B / 64, 256, 0, stream>>>(wf, bf, stm, bswz, ftb, l1w, l1b,
                                        l2w, l2b, l3w, l3b, out, B);
}
